// Backbone_90958817395321
// MI455X (gfx1250) — matrix operand order repaired and checked by reading, not compiled, not run on hardware
//
#include <hip/hip_runtime.h>
#include <math.h>

// ---------------------------------------------------------------------------
// CDNA5 / gfx1250 types
// ---------------------------------------------------------------------------
typedef __attribute__((ext_vector_type(8)))  float        v8f;
typedef __attribute__((ext_vector_type(16))) __bf16       v16bf;
typedef __attribute__((ext_vector_type(4)))  unsigned int u32x4;
typedef __attribute__((ext_vector_type(4)))  unsigned int tdm_u32x4;
typedef __attribute__((ext_vector_type(4)))  int          tdm_i32x4;
typedef __attribute__((ext_vector_type(8)))  int          tdm_i32x8;

union FragB { v16bf v; u32x4 q[2]; };   // 32 bytes: one 16x32 bf16 WMMA operand

#define WMMA_BF16(A, Bm, C) \
  __builtin_amdgcn_wmma_f32_16x16x32_bf16(false, (A), false, (Bm), (short)0, (C), false, false)

// ---------------------------------------------------------------------------
// Tensor Data Mover: async 2D bf16 tile load Global -> LDS.
// Descriptor per CDNA5 ISA (D# groups 0/1; 2D so
// groups 2/3 are zero).  data_size=1 (2-byte elements), type=2 ("image").
// Issued by a single wave; completion via TENSORcnt.
// ---------------------------------------------------------------------------
__device__ __forceinline__ void tdm_load_tile_2d(
    unsigned lds_off,            // byte offset of destination in LDS
    const void* gaddr,           // global address of tile start
    unsigned tensor_d0,          // tensor dim0 length (elements)
    unsigned tensor_d1,          // tensor dim1 length (rows)
    unsigned tile_d0,            // tile dim0 (elements, contiguous)
    unsigned tile_d1,            // tile dim1 (rows)
    unsigned d0_stride)          // dim0 stride (elements)
{
  const unsigned long long ga = (unsigned long long)gaddr;
  tdm_u32x4 g0;
  g0[0] = 1u;                                            // count=1, user descriptor
  g0[1] = lds_off;                                       // lds_addr
  g0[2] = (unsigned)ga;                                  // global_addr[31:0]
  g0[3] = (unsigned)((ga >> 32) & 0x1FFFFFFu) | (2u << 30);  // addr[56:32] | type=2

  tdm_i32x8 g1;
  g1[0] = (int)(1u << 16);                               // wg_mask=0, data_size=1 (2B)
  g1[1] = (int)((tensor_d0 & 0xFFFFu) << 16);            // dim0[15:0] -> bits 63:48
  g1[2] = (int)((tensor_d0 >> 16) | ((tensor_d1 & 0xFFFFu) << 16)); // dim0[31:16], dim1[15:0]
  g1[3] = (int)((tensor_d1 >> 16) | (tile_d0 << 16));    // dim1[31:16], tile_dim0
  g1[4] = (int)(tile_d1 & 0xFFFFu);                      // tile_dim1, tile_dim2=0
  g1[5] = (int)d0_stride;                                // dim0_stride[31:0]
  g1[6] = 0;                                             // dim0_stride[47:32], dim1_stride[15:0]
  g1[7] = 0;                                             // dim1_stride[47:16]

  tdm_i32x4 z4; z4[0] = 0; z4[1] = 0; z4[2] = 0; z4[3] = 0;
#if __clang_major__ >= 23
  tdm_i32x8 z8; for (int i = 0; i < 8; i++) z8[i] = 0;
  __builtin_amdgcn_tensor_load_to_lds(g0, g1, z4, z4, z8, 0);
#else
  __builtin_amdgcn_tensor_load_to_lds(g0, g1, z4, z4, 0);
#endif
}

// ---------------------------------------------------------------------------
// Generic bf16 GEMM:  C[M,N] = act( A[M,K] * B[K,N] + bias )
// A row-major bf16, B row-major (K,N) bf16, f32 accumulate via WMMA.
// Tile: 128(M) x 64(N) per 256-thread workgroup (8 waves, wave = 32x32).
// A tile staged via global_load_b128 -> ds_store_b128 (+ prefetch);
// B tile staged via the Tensor Data Mover (async, TENSORcnt).
// K step = 32.  M%128==0, N%64==0, K%32==0 guaranteed by caller.
// act: 0 = none, 1 = exact GELU.
// ---------------------------------------------------------------------------
__global__ __launch_bounds__(256) void k_gemm(
    const __bf16* __restrict__ A, const __bf16* __restrict__ Bm,
    const float* __restrict__ bias,
    float* __restrict__ Cf, __bf16* __restrict__ Cb,
    int M, int N, int K, int act)
{
  __shared__ __attribute__((aligned(16))) __bf16 As[128 * 32];
  __shared__ __attribute__((aligned(16))) __bf16 Bs[32 * 64];

  const int tid  = threadIdx.x;
  const int lane = tid & 31;
  const int wave = tid >> 5;
  const int wm   = wave >> 1;        // 0..3  (M direction)
  const int wn   = wave & 1;         // 0..1  (N direction)
  const int m16  = lane & 15;
  const int half = lane >> 4;
  const int bm = blockIdx.y, bn = blockIdx.x;

  v8f acc[2][2];
  #pragma unroll
  for (int i = 0; i < 2; i++)
    #pragma unroll
    for (int j = 0; j < 2; j++)
      #pragma unroll
      for (int e = 0; e < 8; e++) acc[i][j][e] = 0.f;

  const int arow = tid >> 1, acol = (tid & 1) * 16;  // A tile 128x32, 16 elems/thread
  const unsigned bs_lds = (unsigned)(unsigned long long)(void*)Bs;

  for (int kb = 0; kb < K; kb += 32) {
    // --- B tile via TDM: 32 rows x 64 cols bf16 from (K,N) weight matrix ---
    if (tid < 32) {
      tdm_load_tile_2d(bs_lds, Bm + (size_t)kb * N + bn * 64,
                       (unsigned)N, (unsigned)K, 64u, 32u, (unsigned)N);
    }
    // --- A tile via vector loads ---
    const u32x4* ag = (const u32x4*)(A + (size_t)(bm * 128 + arow) * K + kb + acol);
    u32x4* as = (u32x4*)(As + arow * 32 + acol);
    as[0] = ag[0]; as[1] = ag[1];
    if (kb + 32 < K) {   // prefetch next A tile (global_prefetch_b8 path)
      __builtin_prefetch(A + (size_t)(bm * 128 + arow) * K + kb + 32 + acol, 0, 1);
    }
    if (tid < 32) __builtin_amdgcn_s_wait_tensorcnt(0);
    __syncthreads();

    FragB a[2], b[2];
    #pragma unroll
    for (int ms = 0; ms < 2; ms++) {
      const u32x4* rp = (const u32x4*)(As + (wm * 32 + ms * 16 + m16) * 32);
      a[ms].q[0] = rp[half];       // k = half*8 .. +7
      a[ms].q[1] = rp[2 + half];   // k = 16+half*8 .. +7
    }
    #pragma unroll
    for (int ns = 0; ns < 2; ns++) {
      #pragma unroll
      for (int s = 0; s < 16; s++)
        b[ns].v.s[0] = b[ns].v.s[0], b[ns].v[s] = Bs[((s & 7) + half * 8 + ((s >> 3) << 4)) * 64 + wn * 32 + ns * 16 + m16];
    }
    #pragma unroll
    for (int ms = 0; ms < 2; ms++)
      #pragma unroll
      for (int ns = 0; ns < 2; ns++)
        acc[ms][ns] = WMMA_BF16(a[ms].v, b[ns].v, acc[ms][ns]);
    __syncthreads();
  }

  // Epilogue.  C/D layout: VGPR r -> row (r + 8*half), lane -> col (lane&15).
  #pragma unroll
  for (int ms = 0; ms < 2; ms++) {
    #pragma unroll
    for (int ns = 0; ns < 2; ns++) {
      const int ncol = bn * 64 + wn * 32 + ns * 16 + m16;
      const float bv = bias ? bias[ncol] : 0.f;
      #pragma unroll
      for (int r = 0; r < 8; r++) {
        const int mrow = bm * 128 + wm * 32 + ms * 16 + r + half * 8;
        float v = acc[ms][ns][r] + bv;
        if (act == 1) v = 0.5f * v * (1.f + erff(v * 0.70710678118f));
        const size_t o = (size_t)mrow * N + ncol;
        if (Cf) Cf[o] = v;
        if (Cb) Cb[o] = (__bf16)v;
      }
    }
  }
}

// ---------------------------------------------------------------------------
// Fused flash-style multi-head attention (D_HEAD = 64, H = 8).
// Q,K,V: (B,H,L,64) bf16.  Output ctx: (B, L, H*64) bf16.
// Block = 128 threads (4 waves); wave handles a 16-query tile; keys in
// blocks of 32; online softmax; scores + P*V via WMMA bf16.
// ---------------------------------------------------------------------------
template <bool CAUSAL>
__global__ __launch_bounds__(128) void k_attn(
    const __bf16* __restrict__ Q, const __bf16* __restrict__ Km,
    const __bf16* __restrict__ Vm, __bf16* __restrict__ ctx,
    int L, int S, int H)
{
  __shared__ __attribute__((aligned(16))) __bf16 ldsp[4][16 * 32];

  const int bh = blockIdx.y;
  const int b = bh / H, h = bh - b * H;
  const int wave = threadIdx.x >> 5, lane = threadIdx.x & 31;
  const int m16 = lane & 15, half = lane >> 4;
  const int qbase = blockIdx.x * 64 + wave * 16;

  const __bf16* Qp = Q  + ((size_t)bh * L + qbase) * 64;
  const __bf16* Kp = Km + (size_t)bh * S * 64;
  const __bf16* Vp = Vm + (size_t)bh * S * 64;

  // Q fragments: chan 0..31 and 32..63 (A layout: lane=M, VGPR=K pairs)
  FragB aq0, aq1;
  {
    const u32x4* rp = (const u32x4*)(Qp + m16 * 64);
    aq0.q[0] = rp[half];     aq0.q[1] = rp[2 + half];
    aq1.q[0] = rp[4 + half]; aq1.q[1] = rp[6 + half];
  }

  float m_i[8], l_i[8];
  v8f o0, o1, o2, o3;
  #pragma unroll
  for (int r = 0; r < 8; r++) {
    m_i[r] = -1e30f; l_i[r] = 0.f;
    o0[r] = 0.f; o1[r] = 0.f; o2[r] = 0.f; o3[r] = 0.f;
  }

  const int blockQHi = blockIdx.x * 64 + 64;
  const int kend = CAUSAL ? (S < blockQHi ? S : blockQHi) : S;   // uniform per block

  for (int kb = 0; kb < kend; kb += 32) {
    v8f s0, s1;
    #pragma unroll
    for (int r = 0; r < 8; r++) { s0[r] = 0.f; s1[r] = 0.f; }

    // scores(16q x 32k) = Q(16x64) * K^T : B operand B[c][n] = K[key n][chan c]
    #pragma unroll
    for (int nt = 0; nt < 2; nt++) {
      #pragma unroll
      for (int ct = 0; ct < 2; ct++) {
        FragB bk;
        {
          const u32x4* kr = (const u32x4*)(Kp + (size_t)(kb + nt * 16 + m16) * 64 + ct * 32);
          bk.q[0] = kr[half]; bk.q[1] = kr[2 + half];
        }
        if (nt == 0) s0 = WMMA_BF16(ct ? aq1.v : aq0.v, bk.v, s0);
        else         s1 = WMMA_BF16(ct ? aq1.v : aq0.v, bk.v, s1);
      }
    }

    // online softmax over this 32-key block
    const float scale = 0.125f;  // 1/sqrt(64)
    #pragma unroll
    for (int r = 0; r < 8; r++) {
      float x0 = s0[r] * scale, x1 = s1[r] * scale;
      if (CAUSAL) {
        const int q = qbase + r + half * 8;
        if (kb + m16 > q)      x0 = -1e30f;
        if (kb + 16 + m16 > q) x1 = -1e30f;
      }
      float mx = fmaxf(x0, x1);
      #pragma unroll
      for (int off = 1; off < 16; off <<= 1) mx = fmaxf(mx, __shfl_xor(mx, off, 32));
      const float mnew = fmaxf(m_i[r], mx);
      const float p0 = __expf(x0 - mnew), p1 = __expf(x1 - mnew);
      float rs = p0 + p1;
      #pragma unroll
      for (int off = 1; off < 16; off <<= 1) rs += __shfl_xor(rs, off, 32);
      const float alpha = __expf(m_i[r] - mnew);
      l_i[r] = l_i[r] * alpha + rs;
      m_i[r] = mnew;
      o0[r] *= alpha; o1[r] *= alpha; o2[r] *= alpha; o3[r] *= alpha;
      ldsp[wave][(r + half * 8) * 32 + m16]      = (__bf16)p0;
      ldsp[wave][(r + half * 8) * 32 + 16 + m16] = (__bf16)p1;
    }
    __syncthreads();

    // O(16x64) += P(16x32) * V(32x64)
    FragB ap;
    {
      const u32x4* rp = (const u32x4*)(&ldsp[wave][m16 * 32]);
      ap.q[0] = rp[half]; ap.q[1] = rp[2 + half];
    }
    #pragma unroll
    for (int t = 0; t < 4; t++) {
      FragB bv;
      #pragma unroll
      for (int s = 0; s < 16; s++)
        bv.v[s] = Vp[(size_t)(kb + (s & 7) + half * 8 + ((s >> 3) << 4)) * 64 + t * 16 + m16];
      if (t == 0)      o0 = WMMA_BF16(ap.v, bv.v, o0);
      else if (t == 1) o1 = WMMA_BF16(ap.v, bv.v, o1);
      else if (t == 2) o2 = WMMA_BF16(ap.v, bv.v, o2);
      else             o3 = WMMA_BF16(ap.v, bv.v, o3);
    }
    __syncthreads();
  }

  #pragma unroll
  for (int r = 0; r < 8; r++) {
    const float inv = 1.f / l_i[r];
    const int q = qbase + r + half * 8;
    __bf16* cp = ctx + ((size_t)b * L + q) * (H * 64) + h * 64;
    cp[m16]      = (__bf16)(o0[r] * inv);
    cp[16 + m16] = (__bf16)(o1[r] * inv);
    cp[32 + m16] = (__bf16)(o2[r] * inv);
    cp[48 + m16] = (__bf16)(o3[r] * inv);
  }
}

// ---------------------------------------------------------------------------
// Elementwise / helper kernels
// ---------------------------------------------------------------------------

// DataEmbedding: circular conv1d(k=3, Cin=7) + sinusoid pos emb + mark proj.
__global__ void k_embed(const float* __restrict__ x, const float* __restrict__ mark,
                        const float* __restrict__ tokw, const float* __restrict__ tempw,
                        float* __restrict__ Xf, __bf16* __restrict__ Xb, int L)
{
  const int idx = blockIdx.x * blockDim.x + threadIdx.x;
  const int total = 8 * L * 512;
  if (idx >= total) return;
  const int d = idx & 511;
  const int bt = idx >> 9;
  const int t = bt % L;
  const int b = bt / L;

  float acc = 0.f;
  #pragma unroll
  for (int j = 0; j < 3; j++) {
    int tt = t + j - 1;
    if (tt < 0) tt += L; else if (tt >= L) tt -= L;   // wrap pad
    const float* xp = x + ((size_t)b * L + tt) * 7;
    const float* wp = tokw + (size_t)d * 21 + j;      // w[d][c][j], stride 3 over c
    #pragma unroll
    for (int c = 0; c < 7; c++) acc += xp[c] * wp[c * 3];
  }
  const float* mp = mark + ((size_t)b * L + t) * 4;
  const float* tw = tempw + (size_t)d * 4;
  #pragma unroll
  for (int m = 0; m < 4; m++) acc += mp[m] * tw[m];

  const int i2 = d >> 1;
  const float freq = __expf(-logf(10000.f) * (float)(2 * i2) * (1.f / 512.f));
  const float ang = (float)t * freq;
  acc += (d & 1) ? __cosf(ang) : __sinf(ang);

  Xf[idx] = acc;
  Xb[idx] = (__bf16)acc;
}

// LayerNorm of (x [+ res]) over last dim 512; emits f32 + bf16. In-place safe.
__global__ __launch_bounds__(256) void k_addln(
    const float* __restrict__ x, const float* __restrict__ res,
    const float* __restrict__ g, const float* __restrict__ bb,
    float* __restrict__ of, __bf16* __restrict__ ob)
{
  __shared__ float s1[256], s2[256];
  const int row = blockIdx.x, tid = threadIdx.x;
  const float* xp = x + (size_t)row * 512;
  const float* rp = res ? res + (size_t)row * 512 : nullptr;
  float v0 = xp[tid]       + (rp ? rp[tid]       : 0.f);
  float v1 = xp[tid + 256] + (rp ? rp[tid + 256] : 0.f);
  s1[tid] = v0 + v1;
  s2[tid] = v0 * v0 + v1 * v1;
  __syncthreads();
  for (int s = 128; s > 0; s >>= 1) {
    if (tid < s) { s1[tid] += s1[tid + s]; s2[tid] += s2[tid + s]; }
    __syncthreads();
  }
  const float mean = s1[0] * (1.f / 512.f);
  const float var  = s2[0] * (1.f / 512.f) - mean * mean;
  const float rstd = rsqrtf(var + 1e-5f);
  const float y0 = (v0 - mean) * rstd * g[tid]       + bb[tid];
  const float y1 = (v1 - mean) * rstd * g[tid + 256] + bb[tid + 256];
  of[(size_t)row * 512 + tid]       = y0;
  of[(size_t)row * 512 + tid + 256] = y1;
  if (ob) {
    ob[(size_t)row * 512 + tid]       = (__bf16)y0;
    ob[(size_t)row * 512 + tid + 256] = (__bf16)y1;
  }
}

// (B,L,512) f32 -> (B,H,L,64) bf16 head split
__global__ void k_split(const float* __restrict__ in, __bf16* __restrict__ out, int L)
{
  const int idx = blockIdx.x * blockDim.x + threadIdx.x;
  const int total = 8 * L * 512;
  if (idx >= total) return;
  const int d = idx & 511;
  const int l = (idx >> 9) % L;
  const int b = idx / (L * 512);
  const int h = d >> 6, c = d & 63;
  out[(((size_t)b * 8 + h) * L + l) * 64 + c] = (__bf16)in[idx];
}

// Distilling stage: BN-scale + ELU + maxpool3 stride2 (-inf pad).
__global__ void k_pool(const float* __restrict__ y, const float* __restrict__ g,
                       const float* __restrict__ bb,
                       float* __restrict__ of, __bf16* __restrict__ ob, int Lin)
{
  const int Lo = Lin >> 1;
  const int idx = blockIdx.x * blockDim.x + threadIdx.x;
  const int total = 8 * Lo * 512;
  if (idx >= total) return;
  const int c = idx & 511;
  const int t = (idx >> 9) % Lo;
  const int b = idx / (Lo * 512);
  const float sg = g[c] * rsqrtf(1.f + 1e-5f);
  float best = -1e30f;
  #pragma unroll
  for (int j = -1; j <= 1; j++) {
    const int tt = 2 * t + j;
    if (tt < 0 || tt >= Lin) continue;
    const float v = y[((size_t)b * Lin + tt) * 512 + c] * sg + bb[c];
    const float e = v > 0.f ? v : expm1f(v);
    best = fmaxf(best, e);
  }
  of[idx] = best;
  ob[idx] = (__bf16)best;
}

// im2col for k=3 circular conv: (B,L,512) f32 -> (B*L, 1536) bf16, K index = c*3+j
__global__ void k_im2col(const float* __restrict__ X, __bf16* __restrict__ col, int L)
{
  const int idx = blockIdx.x * blockDim.x + threadIdx.x;
  const int total = 8 * L * 1536;
  if (idx >= total) return;
  const int k = idx % 1536;
  const int t = (idx / 1536) % L;
  const int b = idx / (1536 * L);
  const int c = k / 3, j = k - 3 * c;
  int tt = t + j - 1;
  if (tt < 0) tt += L; else if (tt >= L) tt -= L;
  col[idx] = (__bf16)X[((size_t)b * L + tt) * 512 + c];
}

// weight transpose: (Nout, Kin) f32 -> (Kin, Nout) bf16
__global__ void k_prep_lin(const float* __restrict__ w, __bf16* __restrict__ out, int N, int Kin)
{
  const int idx = blockIdx.x * blockDim.x + threadIdx.x;
  if (idx >= N * Kin) return;
  const int n = idx % N, k = idx / N;
  out[idx] = (__bf16)w[(size_t)n * Kin + k];
}

// conv3 weight: (512,512,3) f32 -> (1536, 512) bf16 with K = c*3+j
__global__ void k_prep_conv3(const float* __restrict__ w, __bf16* __restrict__ out)
{
  const int idx = blockIdx.x * blockDim.x + threadIdx.x;
  if (idx >= 512 * 512 * 3) return;
  const int o = idx % 512;
  const int k = idx / 512;
  const int c = k / 3, j = k - 3 * c;
  out[idx] = (__bf16)w[((size_t)o * 512 + c) * 3 + j];
}

// final slice: dec[:, -512:, :]
__global__ void k_slice(const float* __restrict__ X, float* __restrict__ out)
{
  const int idx = blockIdx.x * blockDim.x + threadIdx.x;
  if (idx >= 8 * 512 * 512) return;
  const int d = idx & 511;
  const int t = (idx >> 9) & 511;
  const int b = idx >> 18;
  out[idx] = X[((size_t)b * 1024 + 512 + t) * 512 + d];
}

// ---------------------------------------------------------------------------
// Host orchestration
// ---------------------------------------------------------------------------
// Input order (JAX pytree leaf order: dicts sorted alphabetically):
// 0 x_enc, 1 x_mark_enc, 2 x_dec, 3 x_mark_dec, then params:
//   conv_layers[i]: 4+4i: b, bn_b, bn_g, w
//   dec_emb: 12 temp, 13 tok
//   dec_layers[i] (base 14+26i): c1.b,c1.w,c2.b,c2.w,
//       cattn{k.b,k.w,o.b,o.w,q.b,q.w,v.b,v.w}, n1.b,n1.g,n2.b,n2.g,n3.b,n3.g,
//       sattn{k.b,k.w,o.b,o.w,q.b,q.w,v.b,v.w}
//   dec_norm: 66 b, 67 g
//   enc_emb: 68 temp, 69 tok
//   enc_layers[i] (base 70+16i): attn{k.b,k.w,o.b,o.w,q.b,q.w,v.b,v.w},
//       c1.b,c1.w,c2.b,c2.w, n1.b,n1.g, n2.b,n2.g
//   enc_norm: 118 b, 119 g
extern "C" void kernel_launch(void* const* d_in, const int* in_sizes, int n_in,
                              void* d_out, int out_size, void* d_ws, size_t ws_size,
                              hipStream_t stream)
{
  (void)in_sizes; (void)n_in; (void)out_size; (void)ws_size;
  const int B = 8, H = 8;

  char* wsb = (char*)d_ws;
  size_t off = 0;
  auto alloc = [&](size_t bytes) -> void* {
    void* p = wsb + off;
    off += (bytes + 255) & ~(size_t)255;
    return p;
  };

  float*  Xf  = (float*) alloc((size_t)8 * 1024 * 512 * 4);
  __bf16* Xb  = (__bf16*)alloc((size_t)8 * 1024 * 512 * 2);
  float*  Rf  = (float*) alloc((size_t)8 * 1024 * 512 * 4);
  float*  Pf  = (float*) alloc((size_t)8 * 1024 * 512 * 4);
  __bf16* Yb  = (__bf16*)alloc((size_t)8 * 1024 * 2048 * 2);
  __bf16* Qh  = (__bf16*)alloc((size_t)8 * 8 * 1024 * 64 * 2);
  __bf16* Kh  = (__bf16*)alloc((size_t)8 * 8 * 1024 * 64 * 2);
  __bf16* Vh  = (__bf16*)alloc((size_t)8 * 8 * 1024 * 64 * 2);
  __bf16* Cxb = (__bf16*)alloc((size_t)8 * 1024 * 512 * 2);
  float*  Ef  = (float*) alloc((size_t)8 * 256 * 512 * 4);
  __bf16* Eb  = (__bf16*)alloc((size_t)8 * 256 * 512 * 2);
  __bf16* Col = (__bf16*)alloc((size_t)8 * 1024 * 1536 * 2);

  __bf16* WT[120] = {nullptr};
  auto prep_lin = [&](int idx, int Nout, int Kin) {
    __bf16* p = (__bf16*)alloc((size_t)Nout * Kin * 2);
    const int n = Nout * Kin;
    k_prep_lin<<<(n + 255) / 256, 256, 0, stream>>>((const float*)d_in[idx], p, Nout, Kin);
    WT[idx] = p;
  };
  auto prep_conv3 = [&](int idx) {
    __bf16* p = (__bf16*)alloc((size_t)1536 * 512 * 2);
    k_prep_conv3<<<(512 * 512 * 3 + 255) / 256, 256, 0, stream>>>((const float*)d_in[idx], p);
    WT[idx] = p;
  };

  for (int i = 0; i < 3; i++) {            // encoder layers
    const int ba = 70 + i * 16;
    prep_lin(ba + 1, 512, 512);  prep_lin(ba + 3, 512, 512);
    prep_lin(ba + 5, 512, 512);  prep_lin(ba + 7, 512, 512);
    prep_lin(ba + 9, 2048, 512); prep_lin(ba + 11, 512, 2048);
  }
  for (int i = 0; i < 2; i++) {            // decoder layers
    const int ba = 14 + i * 26;
    prep_lin(ba + 1, 2048, 512); prep_lin(ba + 3, 512, 2048);
    prep_lin(ba + 5, 512, 512);  prep_lin(ba + 7, 512, 512);
    prep_lin(ba + 9, 512, 512);  prep_lin(ba + 11, 512, 512);
    prep_lin(ba + 19, 512, 512); prep_lin(ba + 21, 512, 512);
    prep_lin(ba + 23, 512, 512); prep_lin(ba + 25, 512, 512);
  }
  prep_conv3(7);                           // conv_layers[0].w
  prep_conv3(11);                          // conv_layers[1].w

  auto gemm = [&](const __bf16* A, int widx, int bidx, float* Cf2, __bf16* Cb2,
                  int M, int N, int K, int act) {
    dim3 g(N / 64, M / 128);
    k_gemm<<<g, 256, 0, stream>>>(A, WT[widx],
        bidx >= 0 ? (const float*)d_in[bidx] : nullptr, Cf2, Cb2, M, N, K, act);
  };
  auto addln = [&](const float* x, const float* res, int ib, int ig,
                   float* of, __bf16* ob, int rows) {
    k_addln<<<rows, 256, 0, stream>>>(x, res,
        (const float*)d_in[ig], (const float*)d_in[ib], of, ob);
  };
  auto attn = [&](const __bf16* qsrc, int Lq, const __bf16* kv, int Lkv,
                  int qw, int qb, int kw2, int kb2, int vw, int vb,
                  int ow, int ob2, bool causal, float* outf) {
    const int Mq = B * Lq, Mk = B * Lkv;
    gemm(qsrc, qw, qb, Pf, nullptr, Mq, 512, 512, 0);
    k_split<<<(Mq * 512 + 255) / 256, 256, 0, stream>>>(Pf, Qh, Lq);
    gemm(kv, kw2, kb2, Pf, nullptr, Mk, 512, 512, 0);
    k_split<<<(Mk * 512 + 255) / 256, 256, 0, stream>>>(Pf, Kh, Lkv);
    gemm(kv, vw, vb, Pf, nullptr, Mk, 512, 512, 0);
    k_split<<<(Mk * 512 + 255) / 256, 256, 0, stream>>>(Pf, Vh, Lkv);
    dim3 g(Lq / 64, B * H);
    if (causal) k_attn<true><<<g, 128, 0, stream>>>(Qh, Kh, Vh, Cxb, Lq, Lkv, H);
    else        k_attn<false><<<g, 128, 0, stream>>>(Qh, Kh, Vh, Cxb, Lq, Lkv, H);
    gemm(Cxb, ow, ob2, outf, nullptr, Mq, 512, 512, 0);
  };

  // ------------------------------- Encoder -------------------------------
  k_embed<<<(8 * 1024 * 512 + 255) / 256, 256, 0, stream>>>(
      (const float*)d_in[0], (const float*)d_in[1],
      (const float*)d_in[69], (const float*)d_in[68], Xf, Xb, 1024);

  int L = 1024;
  for (int i = 0; i < 3; i++) {
    const int ba = 70 + i * 16;
    const int M = B * L;
    // attention (dense; ProbSparse sampling uses JAX PRNG — approximated dense)
    attn(Xb, L, Xb, L, ba + 5, ba + 4, ba + 1, ba + 0, ba + 7, ba + 6,
         ba + 3, ba + 2, false, Rf);
    addln(Xf, Rf, ba + 12, ba + 13, Xf, Xb, M);                   // n1
    gemm(Xb, ba + 9, ba + 8, nullptr, Yb, M, 2048, 512, 1);       // c1 + GELU
    gemm(Yb, ba + 11, ba + 10, Rf, nullptr, M, 512, 2048, 0);     // c2
    addln(Xf, Rf, ba + 14, ba + 15, Xf, Xb, M);                   // n2
    if (i < 2) {
      const int cb = 4 + i * 4;
      k_im2col<<<(M * 1536 + 255) / 256, 256, 0, stream>>>(Xf, Col, L);
      gemm(Col, cb + 3, cb + 0, Rf, nullptr, M, 512, 1536, 0);
      k_pool<<<((M / 2) * 512 + 255) / 256, 256, 0, stream>>>(
          Rf, (const float*)d_in[cb + 2], (const float*)d_in[cb + 1], Xf, Xb, L);
      L >>= 1;
    }
  }
  addln(Xf, nullptr, 118, 119, Ef, Eb, B * L);                    // enc_norm (L=256)

  // ------------------------------- Decoder -------------------------------
  k_embed<<<(8 * 1024 * 512 + 255) / 256, 256, 0, stream>>>(
      (const float*)d_in[2], (const float*)d_in[3],
      (const float*)d_in[13], (const float*)d_in[12], Xf, Xb, 1024);

  for (int i = 0; i < 2; i++) {
    const int ba = 14 + i * 26;
    const int M = B * 1024;
    // causal self-attention (sattn leaves at +18..+25)
    attn(Xb, 1024, Xb, 1024, ba + 23, ba + 22, ba + 19, ba + 18,
         ba + 25, ba + 24, ba + 21, ba + 20, true, Rf);
    addln(Xf, Rf, ba + 12, ba + 13, Xf, Xb, M);                   // n1
    // cross-attention vs encoder memory (cattn leaves at +4..+11)
    attn(Xb, 1024, Eb, 256, ba + 9, ba + 8, ba + 5, ba + 4,
         ba + 11, ba + 10, ba + 7, ba + 6, false, Rf);
    addln(Xf, Rf, ba + 14, ba + 15, Xf, Xb, M);                   // n2
    gemm(Xb, ba + 1, ba + 0, nullptr, Yb, M, 2048, 512, 1);       // c1 + GELU
    gemm(Yb, ba + 3, ba + 2, Rf, nullptr, M, 512, 2048, 0);       // c2
    addln(Xf, Rf, ba + 16, ba + 17, Xf, Xb, M);                   // n3
  }
  addln(Xf, nullptr, 66, 67, Rf, nullptr, B * 1024);              // dec_norm
  k_slice<<<(8 * 512 * 512 + 255) / 256, 256, 0, stream>>>(Rf, (float*)d_out);
}
